// AE_RNN_HInput_61933428408756
// MI455X (gfx1250) — matrix operand order repaired and checked by reading, not compiled, not run on hardware
//
#include <hip/hip_runtime.h>
#include <hip/hip_bf16.h>

typedef _Float16 v16h __attribute__((ext_vector_type(16)));
typedef _Float16 half8 __attribute__((ext_vector_type(8)));
typedef float    v8f  __attribute__((ext_vector_type(8)));

#define WMMA_F16(a, b, c) \
  __builtin_amdgcn_wmma_f32_16x16x32_f16(false, (a), false, (b), (short)0, (c), false, false)

// ---------------- workspace layout (in _Float16 elements) ----------------
// u transposed:  [B=256][T=1024][U=16] f16
// then pre-swizzled weight fragments, each fragment = 512 f16 (32 lanes x 16)
constexpr size_t UTR_ELEMS = 256ull * 1024 * 16;
constexpr size_t OFF_WIH0  = UTR_ELEMS;                      // 24 ntiles x 1 ktile
constexpr size_t OFF_WHH0  = OFF_WIH0 + 24ull * 1 * 512;     // 24 x 4
constexpr size_t OFF_WIH1  = OFF_WHH0 + 24ull * 4 * 512;     // 24 x 4
constexpr size_t OFF_WHH1  = OFF_WIH1 + 24ull * 4 * 512;     // 24 x 4
constexpr size_t OFF_DW1   = OFF_WHH1 + 24ull * 4 * 512;     // 8 x 5 (K=144 pad 160)
constexpr size_t OFF_DW2   = OFF_DW1  + 8ull  * 5 * 512;     // 1 x 4
constexpr size_t OFF_MW1   = OFF_DW2  + 1ull  * 4 * 512;     // 8 x 5
constexpr size_t OFF_MW2   = OFF_MW1  + 8ull  * 5 * 512;     // 8 x 4
constexpr size_t OFF_MW3   = OFF_MW2  + 8ull  * 4 * 512;     // 1 x 4

// ---------------- fast activations (v_exp_f32 + raw v_rcp_f32) ----------------
__device__ __forceinline__ float fast_rcp(float x) { return __builtin_amdgcn_rcpf(x); }

__device__ __forceinline__ float fsigmoid(float x) {
  x = fminf(fmaxf(x, -30.f), 30.f);
  return fast_rcp(1.f + __expf(-x));
}
__device__ __forceinline__ float ftanh(float x) {
  x = fminf(fmaxf(x, -15.f), 15.f);
  return 1.f - 2.f * fast_rcp(1.f + __expf(2.f * x));   // (e-1)/(e+1)
}

// A fragment (16x32 f16) from row-major LDS buffer [16 x stride].
// ISA layout: lanes 0-15 m=0..15 hold K {0..7,16..23}; lanes 16-31 hold K {8..15,24..31}.
__device__ __forceinline__ v16h load_a(const _Float16* base, int stride, int ktile, int lane) {
  int m = lane & 15;
  int kh = (lane >> 4) * 8;
  const _Float16* p = base + m * stride + ktile * 32 + kh;
  half8 lo = *(const half8*)p;         // K = kh+0..7
  half8 hi = *(const half8*)(p + 16);  // K = kh+16..23
  v16h a;
#pragma unroll
  for (int i = 0; i < 8; ++i) { a[i] = lo[i]; a[i + 8] = hi[i]; }
  return a;
}

// B fragment (32x16 f16) from pre-swizzled global: lane reads its 32 contiguous bytes.
// This variant may be hoisted out of the t-loop by the compiler (register-resident weights).
__device__ __forceinline__ v16h load_b(const _Float16* frag, int lane) {
  const half8* p = (const half8*)(frag + lane * 16);
  half8 lo = p[0];
  half8 hi = p[1];
  v16h b;
#pragma unroll
  for (int i = 0; i < 8; ++i) { b[i] = lo[i]; b[i + 8] = hi[i]; }
  return b;
}

// Non-hoistable variant (volatile): re-fetched each use, hits WGP$/L2.
// Used for feed-forward weights to keep VGPR pressure under the 2-waves/SIMD budget.
__device__ __forceinline__ v16h load_b_nv(const _Float16* frag, int lane) {
  const volatile half8* p = (const volatile half8*)(frag + lane * 16);
  half8 lo = p[0];
  half8 hi = p[1];
  v16h b;
#pragma unroll
  for (int i = 0; i < 8; ++i) { b[i] = lo[i]; b[i + 8] = hi[i]; }
  return b;
}

// One GRU layer; wave j owns hidden tile j (16 units), computes r/z/n and h_new.
__device__ __forceinline__ void gru_layer(
    const _Float16* Agi, int strideGi, int ktGi, int ktilesWih,
    const _Float16* Hprev, const _Float16* Fwih, const _Float16* Fwhh,
    _Float16* Hnext, _Float16* extraDst, int extraStride, int extraOff,
    int wave, int lane)
{
  v8f aR = {}, aZ = {}, aNi = {}, aNh = {};
#pragma unroll
  for (int kt = 0; kt < 4; ++kt) {
    if (kt < ktGi) {
      v16h a = load_a(Agi, strideGi, kt, lane);
      aR  = WMMA_F16(a, load_b(Fwih + (size_t)((wave)      * ktilesWih + kt) * 512, lane), aR);
      aZ  = WMMA_F16(a, load_b(Fwih + (size_t)((8  + wave) * ktilesWih + kt) * 512, lane), aZ);
      aNi = WMMA_F16(a, load_b(Fwih + (size_t)((16 + wave) * ktilesWih + kt) * 512, lane), aNi);
    }
  }
#pragma unroll
  for (int kt = 0; kt < 4; ++kt) {
    v16h a = load_a(Hprev, 128, kt, lane);
    aR  = WMMA_F16(a, load_b(Fwhh + (size_t)((wave)      * 4 + kt) * 512, lane), aR);
    aZ  = WMMA_F16(a, load_b(Fwhh + (size_t)((8  + wave) * 4 + kt) * 512, lane), aZ);
    aNh = WMMA_F16(a, load_b(Fwhh + (size_t)((16 + wave) * 4 + kt) * 512, lane), aNh);
  }
  // C/D layout: lane l elem v -> (m = v + 8*(l>>4), n = l&15)
  int n = lane & 15, mh = (lane >> 4) * 8, col = wave * 16 + n;
#pragma unroll
  for (int v = 0; v < 8; ++v) {
    int m = v + mh;
    float r  = fsigmoid(aR[v]);
    float z  = fsigmoid(aZ[v]);
    float nn = ftanh(aNi[v] + r * aNh[v]);
    float hp = (float)Hprev[m * 128 + col];
    float hv = (1.f - z) * nn + z * hp;
    _Float16 hh = (_Float16)hv;
    Hnext[m * 128 + col] = hh;
    extraDst[m * extraStride + extraOff + col] = hh;
  }
}

// One 16x16 output tile of a Linear layer (wave-local ntile). Weights via load_b_nv.
__device__ __forceinline__ void lin_tile(
    const _Float16* A, int strideA, int ktiles,
    const _Float16* F, int ntile, const float* bias, bool relu,
    _Float16* dst, int strideD, int lane)
{
  v8f acc = {};
  for (int kt = 0; kt < ktiles; ++kt)
    acc = WMMA_F16(load_a(A, strideA, kt, lane),
                   load_b_nv(F + (size_t)(ntile * ktiles + kt) * 512, lane), acc);
  int n = lane & 15, mh = (lane >> 4) * 8;
  float bv = bias[ntile * 16 + n];
#pragma unroll
  for (int v = 0; v < 8; ++v) {
    float x = acc[v] + bv;
    if (relu) x = fmaxf(x, 0.f);
    dst[(v + mh) * strideD + ntile * 16 + n] = (_Float16)x;
  }
}

// ---------------- prep kernels ----------------
__global__ void zero_out(float* p) { p[0] = 0.f; }

// u [256][16][1024] f32 -> ws [256][1024][16] f16
__global__ void utr_kernel(const float* __restrict__ u, _Float16* __restrict__ dst) {
  size_t idx = (size_t)blockIdx.x * 256 + threadIdx.x;
  size_t b = idx >> 14;             // /(16*1024)
  size_t rem = idx & 16383;
  size_t c = rem >> 10;             // /1024
  size_t t = rem & 1023;
  dst[(b * 1024 + t) * 16 + c] = (_Float16)u[idx];
}

// Swizzle weight W[N,K] (row-major) into B fragments: frag f = ntile*ktiles + kt,
// element (lane, j): n = ntile*16 + lane%16, k = kt*32 + (lane/16)*8 + (j<8 ? j : 8+j).
__global__ void swz_kernel(const float* __restrict__ W, _Float16* __restrict__ dst,
                           int N, int K, int ktiles) {
  int idx = blockIdx.x * 256 + threadIdx.x;
  int total = (N / 16) * ktiles * 512;
  if (idx >= total) return;
  int f = idx >> 9;
  int r = idx & 511;
  int lane = r >> 4;
  int j = r & 15;
  int ntile = f / ktiles, ktile = f % ktiles;
  int n = ntile * 16 + (lane & 15);
  int k = ktile * 32 + (lane >> 4) * 8 + (j < 8 ? j : 8 + j);
  float v = (k < K) ? W[(size_t)n * K + k] : 0.f;
  dst[idx] = (_Float16)v;
}

// ---------------- main persistent-recurrence kernel ----------------
__global__ __launch_bounds__(256)
__attribute__((amdgpu_waves_per_eu(2)))
void rnn_main(
    const _Float16* __restrict__ ws16, const float* __restrict__ y,
    const float* __restrict__ h0,
    const float* __restrict__ db1, const float* __restrict__ db2,
    const float* __restrict__ mb1, const float* __restrict__ mb2,
    const float* __restrict__ mb3, float* __restrict__ dout)
{
  __shared__ __align__(16) _Float16 sInp[16 * 32];        // [u(16) | x_prev(16)]
  __shared__ __align__(16) _Float16 sH0[2][16 * 128];
  __shared__ __align__(16) _Float16 sH1[2][16 * 128];
  __shared__ __align__(16) _Float16 sDin[16 * 160];       // [u(16) | h1n(128) | pad(16)]
  __shared__ __align__(16) _Float16 sDmid[16 * 128];
  __shared__ __align__(16) _Float16 sMin[16 * 160];       // [x_t(16) | h0n(128) | pad(16)]
  __shared__ __align__(16) _Float16 sM1[16 * 128];
  __shared__ __align__(16) _Float16 sM2[16 * 128];

  const int tid = threadIdx.x;
  const int wave = tid >> 5, lane = tid & 31;
  const int bBase = blockIdx.x * 16;

  const _Float16* u_tr  = ws16;
  const _Float16* Fwih0 = ws16 + OFF_WIH0;
  const _Float16* Fwhh0 = ws16 + OFF_WHH0;
  const _Float16* Fwih1 = ws16 + OFF_WIH1;
  const _Float16* Fwhh1 = ws16 + OFF_WHH1;
  const _Float16* Fdw1  = ws16 + OFF_DW1;
  const _Float16* Fdw2  = ws16 + OFF_DW2;
  const _Float16* Fmw1  = ws16 + OFF_MW1;
  const _Float16* Fmw2  = ws16 + OFF_MW2;
  const _Float16* Fmw3  = ws16 + OFF_MW3;

  // init h, x0, pad columns
  for (int i = tid; i < 16 * 128; i += 256) {
    int b = i >> 7, h = i & 127;
    sH0[0][i] = (_Float16)h0[(size_t)(bBase + b) * 128 + h];
    sH1[0][i] = (_Float16)h0[(size_t)(256 + bBase + b) * 128 + h];
  }
  {
    int b = tid >> 4, c = tid & 15;
    sInp[b * 32 + 16 + c]   = (_Float16)0.f;   // x_prev(t=0) = 0
    sDin[b * 160 + 144 + c] = (_Float16)0.f;   // K pad
    sMin[b * 160 + 144 + c] = (_Float16)0.f;   // K pad
  }
  __syncthreads();

  float loss = 0.f;

  for (int t = 0; t < 1024; ++t) {
    const int cur = t & 1, nxt = cur ^ 1;

    // ---- load u_t (f16, pre-transposed): 256 threads cover the 16x16 tile ----
    {
      int b = tid >> 4, c = tid & 15;
      _Float16 uv = u_tr[((size_t)(bBase + b) * 1024 + t) * 16 + c];
      sInp[b * 32 + c]  = uv;
      sDin[b * 160 + c] = uv;
    }
    __syncthreads();

    // ---- GRU layer 0: inp[16x32] , h0[16x128] ----
    gru_layer(sInp, 32, 1, 1, sH0[cur], Fwih0, Fwhh0,
              sH0[nxt], sMin, 160, 16, wave, lane);
    __syncthreads();

    // ---- GRU layer 1: h0n[16x128] , h1[16x128] ----
    gru_layer(sH0[nxt], 128, 4, 4, sH1[cur], Fwih1, Fwhh1,
              sH1[nxt], sDin, 160, 16, wave, lane);
    __syncthreads();

    // ---- dynn layer 1: d_in[16x160] @ dW1^T -> [16x128] ----
    lin_tile(sDin, 160, 5, Fdw1, wave, db1, false, sDmid, 128, lane);
    __syncthreads();

    // ---- dynn layer 2 (x_t, 16x16): wave 0 only ----
    if (wave == 0) {
      v8f acc = {};
      for (int kt = 0; kt < 4; ++kt)
        acc = WMMA_F16(load_a(sDmid, 128, kt, lane),
                       load_b_nv(Fdw2 + (size_t)kt * 512, lane), acc);
      int n = lane & 15, mh = (lane >> 4) * 8;
      float bv = db2[n];
#pragma unroll
      for (int v = 0; v < 8; ++v) {
        int m = v + mh;
        _Float16 xv = (_Float16)(acc[v] + bv);
        sInp[m * 32 + 16 + n] = xv;   // x_prev for t+1
        sMin[m * 160 + n]     = xv;   // m_in x part
      }
    }
    __syncthreads();

    // ---- menn layer 1 + ReLU ----
    lin_tile(sMin, 160, 5, Fmw1, wave, mb1, true, sM1, 128, lane);
    __syncthreads();

    // ---- menn layer 2 ----
    lin_tile(sM1, 128, 4, Fmw2, wave, mb2, false, sM2, 128, lane);
    __syncthreads();

    // ---- menn layer 3 (y_hat 16x16) + loss: wave 0 only ----
    if (wave == 0) {
      v8f acc = {};
      for (int kt = 0; kt < 4; ++kt)
        acc = WMMA_F16(load_a(sM2, 128, kt, lane),
                       load_b_nv(Fmw3 + (size_t)kt * 512, lane), acc);
      int n = lane & 15, mh = (lane >> 4) * 8;
      float bv = mb3[n];
#pragma unroll
      for (int v = 0; v < 8; ++v) {
        int m = v + mh;
        float yh = acc[v] + bv;
        float yv = y[((size_t)(bBase + m) * 16 + n) * 1024 + t];
        float d = yh - yv;
        loss += d * d;
      }
    }
    // no barrier needed: next-iter writes don't touch wave0's pending reads
  }

  if (wave == 0) {
#pragma unroll
    for (int off = 16; off > 0; off >>= 1)
      loss += __shfl_xor(loss, off, 32);
    if (lane == 0) atomicAdd(dout, loss);
  }
}

// ---------------- launcher ----------------
extern "C" void kernel_launch(void* const* d_in, const int* in_sizes, int n_in,
                              void* d_out, int out_size, void* d_ws, size_t ws_size,
                              hipStream_t stream) {
  (void)in_sizes; (void)n_in; (void)out_size; (void)ws_size;
  const float* u    = (const float*)d_in[0];
  const float* y    = (const float*)d_in[1];
  const float* h0   = (const float*)d_in[2];
  const float* Wih0 = (const float*)d_in[3];
  const float* Whh0 = (const float*)d_in[4];
  const float* Wih1 = (const float*)d_in[5];
  const float* Whh1 = (const float*)d_in[6];
  const float* dW1  = (const float*)d_in[7];
  const float* db1  = (const float*)d_in[8];
  const float* dW2  = (const float*)d_in[9];
  const float* db2  = (const float*)d_in[10];
  const float* mW1  = (const float*)d_in[11];
  const float* mb1  = (const float*)d_in[12];
  const float* mW2  = (const float*)d_in[13];
  const float* mb2  = (const float*)d_in[14];
  const float* mW3  = (const float*)d_in[15];
  const float* mb3  = (const float*)d_in[16];
  float* dout = (float*)d_out;
  _Float16* ws16 = (_Float16*)d_ws;

  zero_out<<<1, 1, 0, stream>>>(dout);

  // u transpose: 256*16*1024 elements
  utr_kernel<<<(256 * 16 * 1024) / 256, 256, 0, stream>>>(u, ws16);

  auto swz = [&](const float* W, size_t off, int N, int K, int kt) {
    int total = (N / 16) * kt * 512;
    swz_kernel<<<(total + 255) / 256, 256, 0, stream>>>(W, ws16 + off, N, K, kt);
  };
  swz(Wih0, OFF_WIH0, 384, 32, 1);
  swz(Whh0, OFF_WHH0, 384, 128, 4);
  swz(Wih1, OFF_WIH1, 384, 128, 4);
  swz(Whh1, OFF_WHH1, 384, 128, 4);
  swz(dW1,  OFF_DW1,  128, 144, 5);
  swz(dW2,  OFF_DW2,   16, 128, 4);
  swz(mW1,  OFF_MW1,  128, 144, 5);
  swz(mW2,  OFF_MW2,  128, 128, 4);
  swz(mW3,  OFF_MW3,   16, 128, 4);

  rnn_main<<<16, 256, 0, stream>>>(ws16, y, h0, db1, db2, mb1, mb2, mb3, dout);
}
